// Parallel_Graphormer_47021301956916
// MI455X (gfx1250) — compile-verified, matrix operand order checked by reading, not run on hardware
//
#include <hip/hip_runtime.h>
#include <hip/hip_bf16.h>

// ---------------------------------------------------------------------------
// Parallel Graphormer block for MI455X (gfx1250, wave32, WMMA).
// All channel GEMMs run through v_wmma_f32_16x16x32_f16; tile staging uses
// GLOBAL_LOAD_ASYNC_TO_LDS_B128 (ASYNCcnt) to bypass VGPRs.
// ---------------------------------------------------------------------------

#define BB   16
#define DIM  128
#define SEQ  8192
#define HEAD 4
#define HID  512
#define NPOS ((size_t)BB * SEQ)

typedef __attribute__((ext_vector_type(16))) _Float16 v16h;
typedef __attribute__((ext_vector_type(8)))  float    v8f;

__device__ __forceinline__ v8f wmma32(v16h a, v16h b, v8f c) {
  // D = A(16x32 f16) * B(32x16 f16) + C(16x16 f32)
  return __builtin_amdgcn_wmma_f32_16x16x32_f16(
      /*neg_a=*/false, a, /*neg_b=*/false, b,
      /*c_mod=*/(short)0, c, /*reuse_a=*/false, /*reuse_b=*/false);
}

// Async global -> LDS 16-byte copy (per active lane). VDST VGPR carries the
// LDS byte address (low 32 bits of the generic shared pointer), VADDR the
// 64-bit global address. Tracked by ASYNCcnt.
__device__ __forceinline__ void async_copy_b128(void* lds_ptr,
                                                const void* gptr) {
  unsigned lds_addr = (unsigned)(size_t)lds_ptr;
  unsigned long long ga = (unsigned long long)(size_t)gptr;
  asm volatile("global_load_async_to_lds_b128 %0, %1, off"
               :: "v"(lds_addr), "v"(ga)
               : "memory");
}

__device__ __forceinline__ void wait_async0() {
  asm volatile("s_wait_asynccnt 0" ::: "memory");
}

// Load 16 f16 as two contiguous 16B chunks (p0[0..7], p0[16..23]) matching the
// ISA 16-bit A/B fragment striping: half h<8 -> K = 8*(lane>=16)+h,
// h>=8 -> K = 16 + 8*(lane>=16) + (h-8).
__device__ __forceinline__ v16h load16h(const _Float16* p0) {
  float4 a = *(const float4*)(p0);
  float4 b = *(const float4*)(p0 + 16);
  v16h r;
  __builtin_memcpy(&r, &a, 16);
  __builtin_memcpy(((char*)&r) + 16, &b, 16);
  return r;
}

// Row-major fragment: base[row*ldk + kb + ...] with contiguous K runs.
__device__ __forceinline__ v16h frag_rm(const _Float16* base, int ldk,
                                        int row, int kb, int lane) {
  return load16h(base + (size_t)row * ldk + kb + ((lane >> 4) << 3));
}

// Gather fragment for strided K (e.g. channel = K*HEAD + head).
__device__ __forceinline__ v16h frag_gather(const _Float16* base, int stride,
                                            int lane) {
  v16h r = {};
  int lo = (lane >> 4) << 3;
  #pragma unroll
  for (int h = 0; h < 16; ++h) {
    int K = h + ((h >= 8) ? 8 : 0) + lo;
    r[h] = base[(size_t)K * stride];
  }
  return r;
}

__device__ __forceinline__ float sigm(float v) {
  return 1.0f / (1.0f + __expf(-v));
}

// ---------------------------------------------------------------------------
// Weight conversion f32 -> f16
// ---------------------------------------------------------------------------
__global__ void k_cvt(const float* __restrict__ s, _Float16* __restrict__ d,
                      int n) {
  int i = blockIdx.x * 256 + threadIdx.x;
  if (i < n) d[i] = (_Float16)s[i];
}

// ---------------------------------------------------------------------------
// RMSNorm over channels + f16 conversion of xn and z.
// Layout in: (b, c, n) f32.  Layout out: (b*n, c) f16 (GEMM-friendly rows).
// ---------------------------------------------------------------------------
__global__ void k_norm(const float* __restrict__ x, const float* __restrict__ z,
                       const float* __restrict__ nw,
                       _Float16* __restrict__ xnh, _Float16* __restrict__ zh) {
  int gid = blockIdx.x * 256 + threadIdx.x;   // 0 .. B*SEQ-1
  int b = gid >> 13;                          // / SEQ
  int p = gid & (SEQ - 1);
  const float* xb = x + ((size_t)b * DIM) * SEQ + p;
  const float* zb = z + ((size_t)b * DIM) * SEQ + p;
  float ss = 0.0f;
  #pragma unroll 8
  for (int c = 0; c < DIM; ++c) {
    float v = xb[(size_t)c * SEQ];
    ss += v * v;
  }
  float rinv = rsqrtf(ss * (1.0f / DIM) + 1e-6f);
  _Float16* xo = xnh + (size_t)gid * DIM;
  _Float16* zo = zh + (size_t)gid * DIM;
  #pragma unroll 8
  for (int c = 0; c < DIM; ++c) {
    xo[c] = (_Float16)(nw[c] * xb[(size_t)c * SEQ] * rinv);
    zo[c] = (_Float16)(zb[(size_t)c * SEQ]);
  }
}

// ---------------------------------------------------------------------------
// Correlation: corr[b,h] += xr(32 x T) @ k(32 x T)^T with k = Kw@xn + Kb.
// One block per (b, 32-position tile), 8 waves.
// ---------------------------------------------------------------------------
__global__ void k_corr(const _Float16* __restrict__ xnh,
                       const _Float16* __restrict__ Kwh,
                       const float* __restrict__ Kb,
                       float* __restrict__ corr) {
  __shared__ __align__(16) _Float16 xs[32 * DIM];   // [pos][c]
  __shared__ __align__(16) _Float16 ks[32 * DIM];   // [pos][c]
  int blk = blockIdx.x;
  int b = blk / (SEQ / 32);
  int pb = (blk % (SEQ / 32)) * 32;
  int tid = threadIdx.x, lane = tid & 31, wave = tid >> 5;
  int row = lane & 15, hi8 = (lane >> 4) << 3;

  // Async-stage the xn tile (8 KB) straight into LDS, bypassing VGPRs.
  const _Float16* src = xnh + ((size_t)b * SEQ + pb) * DIM;
  for (int i = tid; i < (32 * DIM) / 8; i += 256)
    async_copy_b128(xs + i * 8, src + i * 8);
  wait_async0();
  __syncthreads();

  // k = Kw @ xn + Kb : wave owns output chunk [wave*16, wave*16+16)
  #pragma unroll
  for (int nc = 0; nc < 2; ++nc) {
    v8f acc = {};
    #pragma unroll
    for (int kb = 0; kb < 4; ++kb) {
      v16h a = frag_rm(Kwh, DIM, wave * 16 + row, kb * 32, lane);
      v16h bf = frag_rm(xs + nc * 16 * DIM, DIM, row, kb * 32, lane);
      acc = wmma32(a, bf, acc);
    }
    int o0 = wave * 16 + hi8;
    int p = nc * 16 + row;
    #pragma unroll
    for (int r = 0; r < 8; ++r)
      ks[p * DIM + o0 + r] = (_Float16)(acc[r] + Kb[o0 + r]);
  }
  __syncthreads();

  // corr partial: A = xr (32 x 32 pos), B = k^T (32 pos x 32), K = 32 positions
  int head = wave >> 1, mc = wave & 1;
  #pragma unroll
  for (int nc = 0; nc < 2; ++nc) {
    v16h a = frag_gather(xs + ((mc * 16 + row) * HEAD + head), DIM, lane);
    v16h bf = frag_gather(ks + ((nc * 16 + row) * HEAD + head), DIM, lane);
    v8f acc = {};
    acc = wmma32(a, bf, acc);
    float* cp = corr + (((size_t)b * HEAD + head) * 32) * 32;
    int d0 = mc * 16 + hi8;
    int s = nc * 16 + row;
    #pragma unroll
    for (int r = 0; r < 8; ++r) atomicAdd(&cp[(d0 + r) * 32 + s], acc[r]);
  }
}

// ---------------------------------------------------------------------------
// Main per-tile kernel: SwiGLU FFN + Q projection + corr@q + gated combine.
// One block per (b, 16-position tile), 8 waves.
// ---------------------------------------------------------------------------
__global__ void k_main(const _Float16* __restrict__ xnh,
                       const _Float16* __restrict__ zh,
                       const _Float16* __restrict__ W1h,
                       const float* __restrict__ W1b,
                       const _Float16* __restrict__ W2h,
                       const float* __restrict__ W2b,
                       const _Float16* __restrict__ W3h,
                       const float* __restrict__ W3b,
                       const _Float16* __restrict__ Qwh,
                       const float* __restrict__ Qb,
                       const float* __restrict__ corr,
                       const float* __restrict__ aalpha,
                       const float* __restrict__ abeta,
                       const float* __restrict__ galpha,
                       const float* __restrict__ gbeta,
                       float* __restrict__ out) {
  __shared__ __align__(16) _Float16 xs[16 * DIM];        // xn tile [p][c]
  __shared__ __align__(16) _Float16 zs[16 * DIM];        // z  tile [p][c]
  __shared__ __align__(16) _Float16 gh[16 * HID];        // silu(h1)*h2 [p][o]
  __shared__ __align__(16) _Float16 qs[16 * DIM];        // q tile [p][c]
  __shared__ __align__(16) _Float16 cs[HEAD * 32 * 32];  // corr f16 [h][d][s]
  __shared__ __align__(16) float    fs[16 * DIM];        // ffn f32 [p][c]

  int blk = blockIdx.x;
  int b = blk / (SEQ / 16);
  int pb = (blk % (SEQ / 16)) * 16;
  int tid = threadIdx.x, lane = tid & 31, wave = tid >> 5;
  int row = lane & 15, hi8 = (lane >> 4) << 3;

  // Async-stage xn and z tiles (4 KB each) into LDS, bypassing VGPRs.
  const _Float16* sx = xnh + ((size_t)b * SEQ + pb) * DIM;
  const _Float16* sz = zh + ((size_t)b * SEQ + pb) * DIM;
  __builtin_prefetch(sx + 16 * DIM, 0, 0);   // next tile -> global_prefetch_b8
  for (int i = tid; i < (16 * DIM) / 8; i += 256) {
    async_copy_b128(xs + i * 8, sx + i * 8);
    async_copy_b128(zs + i * 8, sz + i * 8);
  }
  // Corr needs an f32->f16 conversion, so it rides the normal VGPR path.
  const float* cb = corr + (size_t)b * HEAD * 32 * 32;
  for (int i = tid; i < HEAD * 32 * 32; i += 256) cs[i] = (_Float16)cb[i];
  wait_async0();
  __syncthreads();

  // ---- W1 / W2 GEMMs (512x128 each), wave owns 4 output chunks of 16 ----
  for (int m = 0; m < 4; ++m) {
    int ob = (wave * 4 + m) * 16;
    v8f a1 = {}, a2 = {};
    #pragma unroll
    for (int kb = 0; kb < 4; ++kb) {
      v16h bf = frag_rm(xs, DIM, row, kb * 32, lane);
      v16h af1 = frag_rm(W1h, DIM, ob + row, kb * 32, lane);
      a1 = wmma32(af1, bf, a1);
      v16h af2 = frag_rm(W2h, DIM, ob + row, kb * 32, lane);
      a2 = wmma32(af2, bf, a2);
    }
    int o0 = ob + hi8, p = row;
    #pragma unroll
    for (int r = 0; r < 8; ++r) {
      float h1 = a1[r] + W1b[o0 + r];
      float h2 = a2[r] + W2b[o0 + r];
      float g = (h1 / (1.0f + __expf(-h1))) * h2;   // silu(h1)*h2
      gh[p * HID + o0 + r] = (_Float16)g;
    }
  }

  // ---- Q projection (128x128), wave owns one output chunk ----
  {
    int ob = wave * 16;
    v8f aq = {};
    #pragma unroll
    for (int kb = 0; kb < 4; ++kb) {
      v16h bf = frag_rm(zs, DIM, row, kb * 32, lane);
      v16h af = frag_rm(Qwh, DIM, ob + row, kb * 32, lane);
      aq = wmma32(af, bf, aq);
    }
    int o0 = ob + hi8, p = row;
    #pragma unroll
    for (int r = 0; r < 8; ++r)
      qs[p * DIM + o0 + r] = (_Float16)(aq[r] + Qb[o0 + r]);
  }
  __syncthreads();

  // ---- W3 GEMM (128x512), wave owns one output chunk, K = 512 ----
  {
    int ob = wave * 16;
    v8f a3 = {};
    #pragma unroll
    for (int kb = 0; kb < 16; ++kb) {
      v16h aw = frag_rm(W3h, HID, ob + row, kb * 32, lane);
      v16h bg = frag_rm(gh, HID, row, kb * 32, lane);
      a3 = wmma32(aw, bg, a3);
    }
    int o0 = ob + hi8, p = row;
    #pragma unroll
    for (int r = 0; r < 8; ++r)
      fs[p * DIM + o0 + r] = a3[r] + W3b[o0 + r];
  }
  __syncthreads();

  // ---- attn = corr @ q, then gated combine + store ----
  {
    int head = wave >> 1, mc = wave & 1;
    v16h ac = frag_rm(cs + head * 32 * 32, 32, mc * 16 + row, 0, lane);
    v16h bq = frag_gather(qs + row * DIM + head, HEAD, lane);
    v8f at = {};
    at = wmma32(ac, bq, at);
    float sa = sigm(aalpha[head]);
    float sb = sigm(abeta[head]);
    float ga = sigm(galpha[0]);
    float gb = sigm(gbeta[0]);
    int p = row, d0 = mc * 16 + hi8;
    #pragma unroll
    for (int r = 0; r < 8; ++r) {
      int d = d0 + r;
      int c = d * HEAD + head;
      float attnv = at[r] * (1.0f / SEQ);
      float mha = sa * (float)xs[p * DIM + c] + sb * attnv;
      out[(((size_t)b * DIM) + c) * SEQ + pb + p] =
          ga * fs[p * DIM + c] + gb * mha;
    }
  }
}

// ---------------------------------------------------------------------------
// Host launch
// ---------------------------------------------------------------------------
extern "C" void kernel_launch(void* const* d_in, const int* in_sizes, int n_in,
                              void* d_out, int out_size, void* d_ws,
                              size_t ws_size, hipStream_t stream) {
  const float* x  = (const float*)d_in[0];
  const float* z  = (const float*)d_in[1];
  const float* nw = (const float*)d_in[2];
  const float* Qw = (const float*)d_in[3];
  const float* Qb = (const float*)d_in[4];
  const float* Kw = (const float*)d_in[5];
  const float* Kb = (const float*)d_in[6];
  const float* aalpha = (const float*)d_in[7];
  const float* abeta  = (const float*)d_in[8];
  const float* W1w = (const float*)d_in[9];
  const float* W1b = (const float*)d_in[10];
  const float* W2w = (const float*)d_in[11];
  const float* W2b = (const float*)d_in[12];
  const float* W3w = (const float*)d_in[13];
  const float* W3b = (const float*)d_in[14];
  const float* galpha = (const float*)d_in[15];
  const float* gbeta  = (const float*)d_in[16];
  float* out = (float*)d_out;

  // Workspace carve-up
  char* p = (char*)d_ws;
  _Float16* xnh = (_Float16*)p; p += NPOS * DIM * 2;
  _Float16* zh  = (_Float16*)p; p += NPOS * DIM * 2;
  _Float16* W1h = (_Float16*)p; p += (size_t)HID * DIM * 2;
  _Float16* W2h = (_Float16*)p; p += (size_t)HID * DIM * 2;
  _Float16* W3h = (_Float16*)p; p += (size_t)DIM * HID * 2;
  _Float16* Qwh = (_Float16*)p; p += (size_t)DIM * DIM * 2;
  _Float16* Kwh = (_Float16*)p; p += (size_t)DIM * DIM * 2;
  float* corr = (float*)p;      p += (size_t)BB * HEAD * 32 * 32 * 4;

  hipMemsetAsync(corr, 0, (size_t)BB * HEAD * 32 * 32 * 4, stream);

  // Weight conversion f32 -> f16
  k_cvt<<<(HID * DIM + 255) / 256, 256, 0, stream>>>(W1w, W1h, HID * DIM);
  k_cvt<<<(HID * DIM + 255) / 256, 256, 0, stream>>>(W2w, W2h, HID * DIM);
  k_cvt<<<(DIM * HID + 255) / 256, 256, 0, stream>>>(W3w, W3h, DIM * HID);
  k_cvt<<<(DIM * DIM + 255) / 256, 256, 0, stream>>>(Qw, Qwh, DIM * DIM);
  k_cvt<<<(DIM * DIM + 255) / 256, 256, 0, stream>>>(Kw, Kwh, DIM * DIM);

  // RMSNorm + f16 conversion
  k_norm<<<(int)(NPOS / 256), 256, 0, stream>>>(x, z, nw, xnh, zh);

  // Per-head 32x32 correlations (reduction over n)
  k_corr<<<BB * (SEQ / 32), 256, 0, stream>>>(xnh, Kwh, Kb, corr);

  // FFN + attention apply + gated combine
  k_main<<<BB * (SEQ / 16), 256, 0, stream>>>(
      xnh, zh, W1h, W1b, W2h, W2b, W3h, W3b, Qwh, Qb, corr,
      aalpha, abeta, galpha, gbeta, out);
}
